// LinGATEncoder_89635967467601
// MI455X (gfx1250) — compile-verified, matrix operand order checked by reading, not run on hardware
//
#include <hip/hip_runtime.h>
#include <hip/hip_bf16.h>

typedef __attribute__((ext_vector_type(16))) _Float16 v16h;
typedef __attribute__((ext_vector_type(8)))  float    v8f;

#define IN_CH     512
#define OUT_CH    64
#define NEG_SLOPE 0.2f

// ---- monotone float <-> uint key (for atomicMax on floats) ----
__device__ __forceinline__ unsigned fkey(float f) {
  unsigned u = __float_as_uint(f);
  return (u & 0x80000000u) ? ~u : (u | 0x80000000u);
}
__device__ __forceinline__ float funkey(unsigned k) {
  unsigned u = (k & 0x80000000u) ? (k & 0x7FFFFFFFu) : ~k;
  return __uint_as_float(u);
}
#define KEY_NEG_INF 0x007FFFFFu  // fkey(-inf)

// CDNA5 async global->LDS copy of 32B per thread (2 x b128), tracked by ASYNCcnt.
__device__ __forceinline__ void async_copy32(unsigned lds_addr, const void* gaddr) {
  asm volatile(
      "global_load_async_to_lds_b128 %0, %1, off\n\t"
      "global_load_async_to_lds_b128 %0, %1, off offset:16"
      :: "v"(lds_addr), "v"(gaddr) : "memory");
}
__device__ __forceinline__ void wait_async0() {
  asm volatile("s_wait_asynccnt 0x0" ::: "memory");
}

// ---- init: out = bias broadcast; segmax = -inf key; segsum = 0 ----
__global__ void k_init(float* __restrict__ out, const float* __restrict__ bias,
                       unsigned* __restrict__ segkey, float* __restrict__ segsum, int n) {
  int idx = blockIdx.x * blockDim.x + threadIdx.x;
  if (idx < n * OUT_CH) out[idx] = bias[idx & (OUT_CH - 1)];
  if (idx < n) { segkey[idx] = KEY_NEG_INF; segsum[idx] = 0.0f; }
}

// ---- pre-swizzle W_l/W_r (f32, [512,64] row-major) into WMMA B-fragment layout ----
// K-step-major so each kk is a contiguous 8KB block for the async LDS copy:
// t = (((kk*2 + m)*4 + ct)*32 + lane); 16 halves (32B) per fragment-lane.
// 16-bit B 32x16 layout (mirrors A): lane<16 -> K {kk*32+0..7, kk*32+16..23},
// lane>=16 -> K {kk*32+8..15, kk*32+24..31}; col = ct*16 + (lane&15).
__global__ void k_wswizzle(const float* __restrict__ Wl, const float* __restrict__ Wr,
                           _Float16* __restrict__ wfrag) {
  int t = blockIdx.x * blockDim.x + threadIdx.x;
  if (t >= 2 * 16 * 4 * 32) return;
  int lane = t & 31;
  int ct   = (t >> 5) & 3;
  int m    = (t >> 7) & 1;
  int kk   = (t >> 8) & 15;
  const float* W = m ? Wr : Wl;
  int col   = ct * 16 + (lane & 15);
  int khalf = lane >> 4;
  _Float16* dst = wfrag + (size_t)t * 16;
  #pragma unroll
  for (int i = 0; i < 8; ++i) {
    int k = kk * 32 + khalf * 8 + i;
    dst[i]     = (_Float16)W[(size_t)k        * OUT_CH + col];
    dst[i + 8] = (_Float16)W[(size_t)(k + 16) * OUT_CH + col];
  }
}

// ---- fused dual GEMM: x_l = x@W_l + b_l, x_r = x@W_r + b_r (WMMA f16 -> f32 acc) ----
// One wave32 computes a 16-row x 64-col tile for BOTH weight matrices.
// B fragments staged in LDS (shared by all 8 waves), double-buffered via
// global_load_async_to_lds_b128 / s_wait_asynccnt.
__global__ void __launch_bounds__(256)
k_gemm(const float* __restrict__ x, const _Float16* __restrict__ wfrag,
       const float* __restrict__ bl, const float* __restrict__ br,
       float* __restrict__ xl, float* __restrict__ xr, int n) {
  __shared__ __align__(32) char smem[2 * 8192];   // 2 buffers x (8 frags x 1KB)

  int tid   = threadIdx.x;
  int lane  = tid & 31;
  int wave  = tid >> 5;
  int tile  = blockIdx.x * 8 + wave;
  int ntile = (n + 15) >> 4;
  // All waves must run the whole loop (cooperative LDS copies + barriers).
  bool valid = tile < ntile;              // wave-uniform
  if (!valid) tile = ntile - 1;

  int lr    = lane & 15;                  // A: row-in-tile ; B/C/D: col-in-tile
  int khalf = lane >> 4;
  int row   = tile * 16 + lr;
  if (row >= n) row = n - 1;
  const float* xrow = x + (size_t)row * IN_CH;

  v8f acc[2][4];
  #pragma unroll
  for (int m = 0; m < 2; ++m)
    #pragma unroll
    for (int ct = 0; ct < 4; ++ct)
      acc[m][ct] = (v8f){0.f, 0.f, 0.f, 0.f, 0.f, 0.f, 0.f, 0.f};

  const char* wbase   = (const char*)wfrag;
  unsigned    lds0    = (unsigned)(uintptr_t)(&smem[0]);
  unsigned    my_lds  = (unsigned)tid * 32u;

  // prologue: stage kk=0 into buffer 0 (8KB spread over 256 threads, 32B each)
  async_copy32(lds0 + my_lds, wbase + (size_t)tid * 32);
  wait_async0();
  __syncthreads();

  for (int kk = 0; kk < 16; ++kk) {
    int cur = kk & 1;
    // kick off next K-step's B fragments into the other buffer
    if (kk + 1 < 16) {
      async_copy32(lds0 + (unsigned)(cur ^ 1) * 8192u + my_lds,
                   wbase + (size_t)(kk + 1) * 8192 + (size_t)tid * 32);
    }

    // A fragment: 32 f32 of this row-chunk -> 16 f16 per lane (ISA 16-bit A layout)
    int kb = kk * 32 + khalf * 8;
    __builtin_prefetch(xrow + kb + 32, 0, 0);
    float4 a0 = *(const float4*)(xrow + kb);
    float4 a1 = *(const float4*)(xrow + kb + 4);
    float4 a2 = *(const float4*)(xrow + kb + 16);
    float4 a3 = *(const float4*)(xrow + kb + 20);
    v16h af;
    af[0]  = (_Float16)a0.x; af[1]  = (_Float16)a0.y; af[2]  = (_Float16)a0.z; af[3]  = (_Float16)a0.w;
    af[4]  = (_Float16)a1.x; af[5]  = (_Float16)a1.y; af[6]  = (_Float16)a1.z; af[7]  = (_Float16)a1.w;
    af[8]  = (_Float16)a2.x; af[9]  = (_Float16)a2.y; af[10] = (_Float16)a2.z; af[11] = (_Float16)a2.w;
    af[12] = (_Float16)a3.x; af[13] = (_Float16)a3.y; af[14] = (_Float16)a3.z; af[15] = (_Float16)a3.w;

    const char* bbase = &smem[cur * 8192];
    #pragma unroll
    for (int m = 0; m < 2; ++m) {
      #pragma unroll
      for (int ct = 0; ct < 4; ++ct) {
        v16h bf = *(const v16h*)(bbase + (((m * 4 + ct) * 32 + lane) * 32));
        acc[m][ct] = __builtin_amdgcn_wmma_f32_16x16x32_f16(
            false, af, false, bf, (short)0, acc[m][ct], false, false);
      }
    }

    if (kk + 1 < 16) {
      wait_async0();      // next buffer resident
      __syncthreads();    // and everyone done reading current buffer
    }
  }

  // C/D layout: VGPR r, lanes 0-15 -> M=r, lanes 16-31 -> M=8+r; N = lane&15.
  #pragma unroll
  for (int ct = 0; ct < 4; ++ct) {
    int col = ct * 16 + lr;
    float bvl = bl[col], bvr = br[col];
    #pragma unroll
    for (int r = 0; r < 8; ++r) {
      int orow = tile * 16 + r + 8 * khalf;
      if (valid && orow < n) {
        xl[(size_t)orow * OUT_CH + col] = acc[0][ct][r] + bvl;
        xr[(size_t)orow * OUT_CH + col] = acc[1][ct][r] + bvr;
      }
    }
  }
}

// ---- per-edge logits: att . LeakyReLU(x_l[src] + x_r[dst]); atomic max per dst ----
// 16 lanes per edge, 4 channels (float4) per lane.
__global__ void k_logits(const float* __restrict__ xl, const float* __restrict__ xr,
                         const int* __restrict__ ei, const float* __restrict__ att,
                         float* __restrict__ logits, unsigned* __restrict__ segkey,
                         int E, int ET) {
  int tid  = blockIdx.x * blockDim.x + threadIdx.x;
  int edge = tid >> 4;
  if (edge >= ET) return;
  int g = tid & 15;
  int s, d;
  if (edge < E) { s = ei[edge]; d = ei[E + edge]; }
  else          { s = d = edge - E; }               // self loop
  int c = g * 4;
  float4 a = *(const float4*)(xl + (size_t)s * OUT_CH + c);
  float4 b = *(const float4*)(xr + (size_t)d * OUT_CH + c);
  float4 w = *(const float4*)(att + c);
  float z, p = 0.f;
  z = a.x + b.x; z = z > 0.f ? z : NEG_SLOPE * z; p += w.x * z;
  z = a.y + b.y; z = z > 0.f ? z : NEG_SLOPE * z; p += w.y * z;
  z = a.z + b.z; z = z > 0.f ? z : NEG_SLOPE * z; p += w.z * z;
  z = a.w + b.w; z = z > 0.f ? z : NEG_SLOPE * z; p += w.w * z;
  #pragma unroll
  for (int off = 8; off; off >>= 1) p += __shfl_xor(p, off, 16);
  if (g == 0) {
    logits[edge] = p;
    atomicMax(segkey + d, fkey(p));
  }
}

// ---- e = exp(logit - max[dst]); accumulate denom ----
__global__ void k_expsum(const int* __restrict__ ei, const unsigned* __restrict__ segkey,
                         float* __restrict__ logits, float* __restrict__ segsum,
                         int E, int ET) {
  int edge = blockIdx.x * blockDim.x + threadIdx.x;
  if (edge >= ET) return;
  int d = (edge < E) ? ei[E + edge] : (edge - E);
  float m  = funkey(segkey[d]);
  float ev = __expf(logits[edge] - m);
  logits[edge] = ev;                    // overwrite logits with exp values
  atomicAdd(segsum + d, ev);
}

// ---- out[dst] += alpha * x_l[src]  (16 lanes/edge, 4 f32 atomics each) ----
__global__ void k_aggregate(const float* __restrict__ xl, const int* __restrict__ ei,
                            const float* __restrict__ logits, const float* __restrict__ segsum,
                            float* __restrict__ out, int E, int ET) {
  int tid  = blockIdx.x * blockDim.x + threadIdx.x;
  int edge = tid >> 4;
  if (edge >= ET) return;
  int g = tid & 15;
  int s, d;
  if (edge < E) { s = ei[edge]; d = ei[E + edge]; }
  else          { s = d = edge - E; }
  float alpha = logits[edge] / fmaxf(segsum[d], 1e-16f);
  int c = g * 4;
  float4 v = *(const float4*)(xl + (size_t)s * OUT_CH + c);
  float* o = out + (size_t)d * OUT_CH + c;
  atomicAdd(o + 0, alpha * v.x);
  atomicAdd(o + 1, alpha * v.y);
  atomicAdd(o + 2, alpha * v.z);
  atomicAdd(o + 3, alpha * v.w);
}

extern "C" void kernel_launch(void* const* d_in, const int* in_sizes, int n_in,
                              void* d_out, int out_size, void* d_ws, size_t ws_size,
                              hipStream_t stream) {
  const float* x    = (const float*)d_in[0];
  const int*   ei   = (const int*)d_in[1];
  const float* Wl   = (const float*)d_in[2];
  const float* bl   = (const float*)d_in[3];
  const float* Wr   = (const float*)d_in[4];
  const float* br   = (const float*)d_in[5];
  const float* att  = (const float*)d_in[6];
  const float* bias = (const float*)d_in[7];
  float* out = (float*)d_out;

  int n  = in_sizes[0] / IN_CH;
  int E  = in_sizes[1] / 2;
  int ET = E + n;                       // edges + self loops

  // workspace layout (~59 MB for N=100000, E=1.6M)
  char* ws = (char*)d_ws;
  float*    xl     = (float*)ws;     ws += (size_t)n * OUT_CH * sizeof(float);
  float*    xr     = (float*)ws;     ws += (size_t)n * OUT_CH * sizeof(float);
  _Float16* wfrag  = (_Float16*)ws;  ws += (size_t)2 * 16 * 4 * 32 * 16 * sizeof(_Float16);
  float*    logits = (float*)ws;     ws += (size_t)ET * sizeof(float);
  unsigned* segkey = (unsigned*)ws;  ws += (size_t)n * sizeof(unsigned);
  float*    segsum = (float*)ws;

  int initN = n * OUT_CH;
  k_init<<<(initN + 255) / 256, 256, 0, stream>>>(out, bias, segkey, segsum, n);
  k_wswizzle<<<(2 * 16 * 4 * 32 + 255) / 256, 256, 0, stream>>>(Wl, Wr, wfrag);

  int ntile = (n + 15) / 16;
  k_gemm<<<(ntile + 7) / 8, 256, 0, stream>>>(x, wfrag, bl, br, xl, xr, n);

  long long lt = (long long)ET * 16;
  k_logits<<<(unsigned)((lt + 255) / 256), 256, 0, stream>>>(xl, xr, ei, att, logits, segkey, E, ET);
  k_expsum<<<(ET + 255) / 256, 256, 0, stream>>>(ei, segkey, logits, segsum, E, ET);
  k_aggregate<<<(unsigned)((lt + 255) / 256), 256, 0, stream>>>(xl, ei, logits, segsum, out, E, ET);
}
